// MultiheadAttentionBlock_87763361726712
// MI455X (gfx1250) — matrix operand order repaired and checked by reading, not compiled, not run on hardware
//
#include <hip/hip_runtime.h>
#include <hip/hip_bf16.h>

// ---------------- problem constants ----------------
constexpr int B_  = 2;
constexpr int S_  = 2048;
constexpr int DM_ = 1024;   // d_model
constexpr int H_  = 16;
constexpr int DK_ = 64;     // d_k
constexpr int MROWS = B_ * S_;      // 4096 rows for the big GEMMs

typedef __attribute__((ext_vector_type(16))) __bf16 bf16x16;
typedef __attribute__((ext_vector_type(8)))  __bf16 bf16x8;
typedef __attribute__((ext_vector_type(8)))  float  f32x8;

// hardware fp32 -> bf16 (RNE) convert
__device__ __forceinline__ __bf16 f2bf(float f) { return (__bf16)f; }

__device__ __forceinline__ f32x8 wmma_bf16(bf16x16 a, bf16x16 b, f32x8 c) {
  // 8 args: (neg_a, A, neg_b, B, c_mod, C, reuse_a, reuse_b)
  return __builtin_amdgcn_wmma_f32_16x16x32_bf16(false, a, false, b, (short)0, c,
                                                 false, false);
}

__device__ __forceinline__ bf16x16 concat8(bf16x8 lo, bf16x8 hi) {
  return __builtin_shufflevector(lo, hi, 0, 1, 2, 3, 4, 5, 6, 7,
                                 8, 9, 10, 11, 12, 13, 14, 15);
}

// ---- A fragment (16x32, MxK) from row-major bf16, two 16B loads ----
// lane<16: M=lane,    K = 0..15
// lane>=16: M=lane-16, K = 16..31
__device__ __forceinline__ bf16x16 load_a(const __bf16* base, int ld,
                                          int row0, int k0, int lane) {
  int m  = lane & 15;
  int kb = (lane < 16) ? 0 : 16;
  const __bf16* p = base + (size_t)(row0 + m) * ld + k0 + kb;
  return concat8(*(const bf16x8*)p, *(const bf16x8*)(p + 8));
}

// ---- B fragment (32x16, KxN): column n of B is ROW (n0+n) of row-major R
// (computes X * R^T). One contiguous 32B load per lane. ----
__device__ __forceinline__ bf16x16 load_bT(const __bf16* R, int ld,
                                           int n0, int k0, int lane) {
  int n  = lane & 15;
  int kb = (lane < 16) ? 0 : 16;
  const __bf16* p = R + (size_t)(n0 + n) * ld + k0 + kb;
  return *(const bf16x16*)p;
}

// ============================================================
// Kernel 0: one-shot fp32 -> bf16 conversion (vectorized 8/thread)
// ============================================================
__global__ void cvt_kernel(const float* __restrict__ src,
                           __bf16* __restrict__ dst, int n) {
  int i = (blockIdx.x * blockDim.x + threadIdx.x) * 8;
  if (i >= n) return;
  f32x8 v = *(const f32x8*)(src + i);
  bf16x8 o;
#pragma unroll
  for (int j = 0; j < 8; ++j) o[j] = f2bf(v[j]);
  *(bf16x8*)(dst + i) = o;
}

// ============================================================
// Kernel 1: fused Q/K/V projections. One wave = 16x64 strip (one head).
//   q,k -> [B,H,S,DK] bf16 ; v -> [B,H,DK,S] bf16 (transposed)
// ============================================================
__global__ void proj_kernel(const __bf16* __restrict__ Qb, const __bf16* __restrict__ Kb,
                            const __bf16* __restrict__ Vb,
                            const __bf16* __restrict__ WQb, const float* __restrict__ bQ,
                            const __bf16* __restrict__ WKb, const float* __restrict__ bK,
                            const __bf16* __restrict__ WVb, const float* __restrict__ bV,
                            __bf16* __restrict__ qh, __bf16* __restrict__ kh,
                            __bf16* __restrict__ vth) {
  const int lane = threadIdx.x & 31;
  const int wave = threadIdx.x >> 5;
  const int NS  = DM_ / 64;            // 16 strips of 64 columns
  const int MT  = MROWS / 16;          // 256 row tiles
  const int per = MT * NS;             // 4096 strips per matrix
  int w   = blockIdx.x * (blockDim.x >> 5) + wave;
  int sel = w / per;                   // 0=Q 1=K 2=V
  int t   = w % per;
  int m0  = (t / NS) * 16;
  int n0  = (t % NS) * 64;

  const __bf16* X    = (sel == 0) ? Qb  : (sel == 1) ? Kb  : Vb;
  const __bf16* W    = (sel == 0) ? WQb : (sel == 1) ? WKb : WVb;
  const float*  bias = (sel == 0) ? bQ  : (sel == 1) ? bK  : bV;

  f32x8 acc0 = {}, acc1 = {}, acc2 = {}, acc3 = {};
  for (int k0 = 0; k0 < DM_; k0 += 32) {
    __builtin_prefetch(X + (size_t)m0 * DM_ + k0 + 64, 0, 1);   // speculative: safe OOB
    __builtin_prefetch(W + (size_t)n0 * DM_ + k0 + 64, 0, 1);
    bf16x16 a = load_a(X, DM_, m0, k0, lane);
    acc0 = wmma_bf16(a, load_bT(W, DM_, n0,      k0, lane), acc0);
    acc1 = wmma_bf16(a, load_bT(W, DM_, n0 + 16, k0, lane), acc1);
    acc2 = wmma_bf16(a, load_bT(W, DM_, n0 + 32, k0, lane), acc2);
    acc3 = wmma_bf16(a, load_bT(W, DM_, n0 + 48, k0, lane), acc3);
  }

  const int nn = lane & 15, hi = lane >> 4;
  const int h  = n0 >> 6;              // a 64-strip is exactly one head
  const int b_ = m0 / S_;
  const int s0 = (m0 % S_) + 8 * hi;   // first of 8 consecutive rows this lane owns

  if (sel < 2) {
    __bf16* out = (sel == 0) ? qh : kh;
    size_t base = (((size_t)b_ * H_ + h) * S_ + s0) * DK_ + nn;
#pragma unroll
    for (int r = 0; r < 8; ++r) {
      size_t ro = base + (size_t)r * DK_;
      out[ro]      = f2bf(acc0[r] + bias[n0 + nn]);
      out[ro + 16] = f2bf(acc1[r] + bias[n0 + 16 + nn]);
      out[ro + 32] = f2bf(acc2[r] + bias[n0 + 32 + nn]);
      out[ro + 48] = f2bf(acc3[r] + bias[n0 + 48 + nn]);
    }
  } else {
    size_t hb = ((size_t)b_ * H_ + h) * DK_;
#pragma unroll
    for (int tt = 0; tt < 4; ++tt) {
      f32x8 a = (tt == 0) ? acc0 : (tt == 1) ? acc1 : (tt == 2) ? acc2 : acc3;
      int   d = tt * 16 + nn;
      float bv = bias[n0 + d];
      bf16x8 o;
#pragma unroll
      for (int r = 0; r < 8; ++r) o[r] = f2bf(a[r] + bv);
      *(bf16x8*)(vth + (hb + d) * S_ + s0) = o;   // 8 consecutive s -> one 16B store
    }
  }
}

// ============================================================
// Kernel 2: flash attention. One wave = (head, 16-query block).
// Mask semantics replicate reference: masked score := 1e-9 (NOT -inf).
// ============================================================
__global__ void attn_kernel(const __bf16* __restrict__ qh,
                            const __bf16* __restrict__ kh,
                            const __bf16* __restrict__ vth,
                            const int*    __restrict__ mask,
                            __bf16*       __restrict__ ctx) {
  __shared__ __bf16 plds[8][16 * 32];    // per-wave 16x32 P staging (8KB/block)
  const int lane = threadIdx.x & 31;
  const int wave = threadIdx.x >> 5;
  int wid = blockIdx.x * (blockDim.x >> 5) + wave;
  const int QB = S_ / 16;                // 128 query blocks
  int qb = wid % QB;
  int bh = wid / QB;                     // 0..31
  int b_ = bh / H_, h = bh % H_;

  const __bf16* qbase = qh  + (size_t)bh * S_ * DK_;
  const __bf16* kbase = kh  + (size_t)bh * S_ * DK_;
  const __bf16* vbase = vth + (size_t)bh * DK_ * S_;
  const int*    mbase = mask + (size_t)b_ * S_ * S_;
  const int q0 = qb * 16;
  const int nn = lane & 15, hi = lane >> 4;

  // Q tile (16 x 64) as two K=32 A-fragments, loaded once.
  bf16x16 qa0 = load_a(qbase, DK_, q0, 0,  lane);
  bf16x16 qa1 = load_a(qbase, DK_, q0, 32, lane);

  f32x8 o0 = {}, o1 = {}, o2 = {}, o3 = {};
  float mrow[8], lrow[8];
#pragma unroll
  for (int r = 0; r < 8; ++r) { mrow[r] = -1e30f; lrow[r] = 0.0f; }

  __bf16* pl = plds[wave];

  for (int j = 0; j < S_; j += 32) {
    // ---- scores: two 16x16 tiles (keys j..j+15 and j+16..j+31) ----
    f32x8 s0 = {}, s1 = {};
    s0 = wmma_bf16(qa0, load_bT(kbase, DK_, j,      0,  lane), s0);
    s0 = wmma_bf16(qa1, load_bT(kbase, DK_, j,      32, lane), s0);
    s1 = wmma_bf16(qa0, load_bT(kbase, DK_, j + 16, 0,  lane), s1);
    s1 = wmma_bf16(qa1, load_bT(kbase, DK_, j + 16, 32, lane), s1);

    // ---- scale + mask (masked -> literal 1e-9, as in reference) ----
#pragma unroll
    for (int r = 0; r < 8; ++r) {
      int qrow = q0 + r + 8 * hi;
      const int* mp = mbase + (size_t)qrow * S_ + j + nn;
      float v0 = s0[r] * 0.125f; if (mp[0]  == 0) v0 = 1e-9f;
      float v1 = s1[r] * 0.125f; if (mp[16] == 0) v1 = 1e-9f;
      s0[r] = v0; s1[r] = v1;
    }

    // ---- online softmax (row reductions across 16 lanes) ----
#pragma unroll
    for (int r = 0; r < 8; ++r) {
      float mj = fmaxf(s0[r], s1[r]);
#pragma unroll
      for (int xm = 1; xm < 16; xm <<= 1) mj = fmaxf(mj, __shfl_xor(mj, xm, 32));
      float mnew  = fmaxf(mrow[r], mj);
      float alpha = __expf(mrow[r] - mnew);
      float p0 = __expf(s0[r] - mnew);
      float p1 = __expf(s1[r] - mnew);
      float ps = p0 + p1;
#pragma unroll
      for (int xm = 1; xm < 16; xm <<= 1) ps += __shfl_xor(ps, xm, 32);
      lrow[r] = lrow[r] * alpha + ps;
      mrow[r] = mnew;
      s0[r] = p0; s1[r] = p1;
      o0[r] *= alpha; o1[r] *= alpha; o2[r] *= alpha; o3[r] *= alpha;
    }

    // ---- stage P (16x32) through LDS: C layout -> A-fragment layout ----
#pragma unroll
    for (int r = 0; r < 8; ++r) {
      int row = r + 8 * hi;
      pl[row * 32 + nn]      = f2bf(s0[r]);
      pl[row * 32 + 16 + nn] = f2bf(s1[r]);
    }
    asm volatile("s_wait_dscnt 0" ::: "memory");  // cross-lane LDS visibility (wave-synchronous)
    bf16x16 pa;
    {
      int m  = lane & 15;
      int kb = (lane < 16) ? 0 : 16;
      const __bf16* pp = pl + m * 32 + kb;
      pa = concat8(*(const bf16x8*)pp, *(const bf16x8*)(pp + 8));
    }

    // ---- O += P * V  (4 d-tiles; B columns = rows of v^T, contiguous 32B) ----
    o0 = wmma_bf16(pa, load_bT(vbase, S_, 0,  j, lane), o0);
    o1 = wmma_bf16(pa, load_bT(vbase, S_, 16, j, lane), o1);
    o2 = wmma_bf16(pa, load_bT(vbase, S_, 32, j, lane), o2);
    o3 = wmma_bf16(pa, load_bT(vbase, S_, 48, j, lane), o3);
  }

  // ---- epilogue: O / l, context in [B,S, h*64+d] bf16 (A for out-proj) ----
#pragma unroll
  for (int r = 0; r < 8; ++r) {
    int qrow = q0 + r + 8 * hi;
    size_t rowoff = ((size_t)b_ * S_ + qrow) * DM_ + (size_t)h * DK_;
    float inv = 1.0f / lrow[r];
    ctx[rowoff + 0  + nn] = f2bf(o0[r] * inv);
    ctx[rowoff + 16 + nn] = f2bf(o1[r] * inv);
    ctx[rowoff + 32 + nn] = f2bf(o2[r] * inv);
    ctx[rowoff + 48 + nn] = f2bf(o3[r] * inv);
  }
}

// ============================================================
// Kernel 3: output projection  out = ctx @ WO^T + bO   (fp32 out)
// One wave = 16x64 strip.
// ============================================================
__global__ void outproj_kernel(const __bf16* __restrict__ ctx,
                               const __bf16* __restrict__ WOb,
                               const float*  __restrict__ bO,
                               float*        __restrict__ out) {
  const int lane = threadIdx.x & 31;
  const int wave = threadIdx.x >> 5;
  const int NS = DM_ / 64;             // 16 strips
  int w  = blockIdx.x * (blockDim.x >> 5) + wave;
  int m0 = (w / NS) * 16;
  int n0 = (w % NS) * 64;

  f32x8 acc0 = {}, acc1 = {}, acc2 = {}, acc3 = {};
  for (int k0 = 0; k0 < DM_; k0 += 32) {
    __builtin_prefetch(ctx + (size_t)m0 * DM_ + k0 + 64, 0, 1);
    __builtin_prefetch(WOb + (size_t)n0 * DM_ + k0 + 64, 0, 1);
    bf16x16 a = load_a(ctx, DM_, m0, k0, lane);
    acc0 = wmma_bf16(a, load_bT(WOb, DM_, n0,      k0, lane), acc0);
    acc1 = wmma_bf16(a, load_bT(WOb, DM_, n0 + 16, k0, lane), acc1);
    acc2 = wmma_bf16(a, load_bT(WOb, DM_, n0 + 32, k0, lane), acc2);
    acc3 = wmma_bf16(a, load_bT(WOb, DM_, n0 + 48, k0, lane), acc3);
  }

  const int nn = lane & 15, hi = lane >> 4;
#pragma unroll
  for (int r = 0; r < 8; ++r) {
    size_t ro = (size_t)(m0 + r + 8 * hi) * DM_ + n0 + nn;
    out[ro]      = acc0[r] + bO[n0 + nn];
    out[ro + 16] = acc1[r] + bO[n0 + 16 + nn];
    out[ro + 32] = acc2[r] + bO[n0 + 32 + nn];
    out[ro + 48] = acc3[r] + bO[n0 + 48 + nn];
  }
}

// ============================================================
extern "C" void kernel_launch(void* const* d_in, const int* in_sizes, int n_in,
                              void* d_out, int out_size, void* d_ws, size_t ws_size,
                              hipStream_t stream) {
  (void)in_sizes; (void)n_in; (void)out_size; (void)ws_size;
  const float* Q    = (const float*)d_in[0];
  const float* K    = (const float*)d_in[1];
  const float* V    = (const float*)d_in[2];
  const int*   mask = (const int*)  d_in[3];
  const float* WQ   = (const float*)d_in[4];
  const float* bQ   = (const float*)d_in[5];
  const float* WK   = (const float*)d_in[6];
  const float* bK   = (const float*)d_in[7];
  const float* WV   = (const float*)d_in[8];
  const float* bV   = (const float*)d_in[9];
  const float* WO   = (const float*)d_in[10];
  const float* bO   = (const float*)d_in[11];

  // workspace carve-up (bf16 elements):
  //   qh | kh | vth | ctx | Qb | Kb | Vb | WQb | WKb | WVb | WOb   (~64 MB)
  const size_t PER = (size_t)B_ * H_ * S_ * DK_;   // 4,194,304
  const size_t WEL = (size_t)DM_ * DM_;            // 1,048,576
  __bf16* qh  = (__bf16*)d_ws;
  __bf16* kh  = qh  + PER;
  __bf16* vth = kh  + PER;
  __bf16* ctx = vth + PER;
  __bf16* Qb  = ctx + PER;
  __bf16* Kb  = Qb  + PER;
  __bf16* Vb  = Kb  + PER;
  __bf16* WQb = Vb  + PER;
  __bf16* WKb = WQb + WEL;
  __bf16* WVb = WKb + WEL;
  __bf16* WOb = WVb + WEL;

  // 0) one-shot fp32 -> bf16 conversions
  cvt_kernel<<<(int)(PER / 8 / 256), 256, 0, stream>>>(Q, Qb, (int)PER);
  cvt_kernel<<<(int)(PER / 8 / 256), 256, 0, stream>>>(K, Kb, (int)PER);
  cvt_kernel<<<(int)(PER / 8 / 256), 256, 0, stream>>>(V, Vb, (int)PER);
  cvt_kernel<<<(int)(WEL / 8 / 256), 256, 0, stream>>>(WQ, WQb, (int)WEL);
  cvt_kernel<<<(int)(WEL / 8 / 256), 256, 0, stream>>>(WK, WKb, (int)WEL);
  cvt_kernel<<<(int)(WEL / 8 / 256), 256, 0, stream>>>(WV, WVb, (int)WEL);
  cvt_kernel<<<(int)(WEL / 8 / 256), 256, 0, stream>>>(WO, WOb, (int)WEL);

  // 1) projections: 3 * 256 * 16 = 12288 strips, 8 waves/block
  proj_kernel<<<12288 / 8, 256, 0, stream>>>(Qb, Kb, Vb, WQb, bQ, WKb, bK,
                                             WVb, bV, qh, kh, vth);
  // 2) attention: 32 heads * 128 query-blocks = 4096 waves
  attn_kernel<<<4096 / 8, 256, 0, stream>>>(qh, kh, vth, mask, ctx);
  // 3) output projection: 256 * 16 = 4096 strips
  outproj_kernel<<<4096 / 8, 256, 0, stream>>>(ctx, WOb, bO, (float*)d_out);
}
